// NequIPEnergyModel_MarioJAX_88768384074455
// MI455X (gfx1250) — matrix operand order repaired and checked by reading, not compiled, not run on hardware
//
#include <hip/hip_runtime.h>

#define N_NODES 30000
#define N_EDGES 480000
#define CCH     32
#define LLAYERS 3
#define NBASIS  8
#define NSPEC   5
#define RMAX    4.0f
#define E_TILES (N_EDGES / 16)

typedef __attribute__((ext_vector_type(16))) _Float16 v16h;
typedef __attribute__((ext_vector_type(8)))  _Float16 v8h;
typedef __attribute__((ext_vector_type(8)))  float    v8f;

__device__ __forceinline__ float silu_f(float x) {
  return x * __builtin_amdgcn_rcpf(1.f + __expf(-x));
}

// ---- fragment loaders ----
// B fragments are pre-packed in memory as frag[fragIdx][lane][0..15] (contiguous
// 32B per lane): element i of the lane's v16h is W[kb*32 + (lane>>4)*16 + i][16*t + (lane&15)].
__device__ __forceinline__ v16h loadBfrag(const _Float16* Wf, int frag, int lane) {
  return *(const v16h*)(Wf + (((size_t)frag * 32 + lane) << 4));
}

// A 16x32 f16 from row-major 16x64 LDS tile: lane holds row m = lane&15;
// its 16 elements are one contiguous 16-half run at K = kb*32 + 16*kh + i.
__device__ __forceinline__ v16h loadA32(const _Float16* H, int kblock, int lane) {
  int m  = lane & 15;
  int kh = lane >> 4;
  const _Float16* p = H + m * 64 + kblock * 32;
  v8h lo = *(const v8h*)(p + 16 * kh);
  v8h hi = *(const v8h*)(p + 16 * kh + 8);
  return __builtin_shufflevector(lo, hi, 0, 1, 2, 3, 4, 5, 6, 7,
                                         8, 9, 10, 11, 12, 13, 14, 15);
}

// ---- prologue: lengths / nhat / f16 bessel basis with envelope ----
__global__ void prep_edges(const float* __restrict__ vec,
                           _Float16* __restrict__ rb16,
                           float* __restrict__ nhat) {
  int e = blockIdx.x * blockDim.x + threadIdx.x;
  if (e >= N_EDGES) return;
  float vx = vec[(size_t)e * 3 + 0] * (1.f / RMAX);
  float vy = vec[(size_t)e * 3 + 1] * (1.f / RMAX);
  float vz = vec[(size_t)e * 3 + 2] * (1.f / RMAX);
  float len = sqrtf(vx * vx + vy * vy + vz * vz);
  float inv = __builtin_amdgcn_rcpf(fmaxf(len, 1e-9f));
  nhat[(size_t)e * 3 + 0] = vx * inv;
  nhat[(size_t)e * 3 + 1] = vy * inv;
  nhat[(size_t)e * 3 + 2] = vz * inv;
  // reference passes the already-/R_MAX length into _radial_basis -> x = len/R_MAX again
  float x  = len * (1.f / RMAX);
  float x2 = x * x;
  float x6 = x2 * x2 * x2;
  float u   = 1.f - 28.f * x6 + 48.f * x6 * x - 21.f * x6 * x2;
  float env = (x < 1.f) ? u : 0.f;
  float scale = sqrtf(2.f / RMAX) * inv * env;
  const float PI = 3.14159265358979323846f;
#pragma unroll
  for (int n = 1; n <= NBASIS; n++)
    rb16[(size_t)e * NBASIS + (n - 1)] = (_Float16)(scale * __sinf((float)n * PI * x));
}

// ---- prologue: fp32 -> f16 conversion into B-fragment-packed layout ----
// packed layout per GEMM: frag = (tile*KB + kb); element ((frag*32+lane)*16 + i)
//   K = kb*32 + (lane>>4)*16 + i ; N = tile*16 + (lane&15)
__global__ void convert_weights(const float* __restrict__ wr1, const float* __restrict__ wr2,
                                const float* __restrict__ wro,
                                _Float16* __restrict__ w1p, _Float16* __restrict__ w2h,
                                _Float16* __restrict__ w3h) {
  int t = blockIdx.x * blockDim.x + threadIdx.x;
  const int P1 = LLAYERS * 2048;    // 4 frags  (4 tiles x KB=1), K pad 8->32
  const int P2 = LLAYERS * 4096;    // 8 frags  (4 tiles x KB=2)
  const int P3 = LLAYERS * 10240;   // 20 frags (10 tiles x KB=2)
  if (t < P1) {
    int l = t / 2048, u = t % 2048;
    int i = u & 15, lane = (u >> 4) & 31, frag = u >> 9;   // frag == tile, kb=0
    int K = ((lane >> 4) << 4) + i;
    int N = frag * 16 + (lane & 15);
    w1p[t] = (_Float16)(K < NBASIS ? wr1[((size_t)l * NBASIS + K) * 64 + N] : 0.f);
  } else if (t < P1 + P2) {
    int q = t - P1;
    int l = q / 4096, u = q % 4096;
    int i = u & 15, lane = (u >> 4) & 31, frag = u >> 9;
    int tt = frag >> 1, kb = frag & 1;
    int K = kb * 32 + ((lane >> 4) << 4) + i;
    int N = tt * 16 + (lane & 15);
    w2h[q] = (_Float16)wr2[((size_t)l * 64 + K) * 64 + N];
  } else if (t < P1 + P2 + P3) {
    int q = t - P1 - P2;
    int l = q / 10240, u = q % 10240;
    int i = u & 15, lane = (u >> 4) & 31, frag = u >> 9;
    int tt = frag >> 1, kb = frag & 1;
    int K = kb * 32 + ((lane >> 4) << 4) + i;
    int N = tt * 16 + (lane & 15);
    w3h[q] = (_Float16)wro[((size_t)l * 64 + K) * 160 + N];
  }
}

__global__ void init_nodes(const float* __restrict__ emb, const int* __restrict__ spec,
                           float* __restrict__ s0) {
  int t = blockIdx.x * blockDim.x + threadIdx.x;
  if (t >= N_NODES * CCH) return;
  int n = t >> 5, c = t & 31;
  s0[t] = emb[spec[n] * CCH + c];
}

// ---- edge kernel: WMMA radial MLP fused with message construction + atomic scatter ----
__global__ __launch_bounds__(256) void edge_kernel(
    int layer,
    const _Float16* __restrict__ rb16, const float* __restrict__ nhat,
    const int* __restrict__ senders, const int* __restrict__ receivers,
    const float* __restrict__ s_in, const float* __restrict__ v_in,
    const _Float16* __restrict__ w1p, const _Float16* __restrict__ w2h,
    const _Float16* __restrict__ w3h,
    float* __restrict__ agg_s, float* __restrict__ agg_v) {
  __shared__ __align__(32) _Float16 w1s[4 * 512];    // 4 frags
  __shared__ __align__(32) _Float16 w2s[8 * 512];    // 8 frags
  __shared__ __align__(32) _Float16 w3s[20 * 512];   // 20 frags
  __shared__ __align__(32) _Float16 hbuf[8 * 16 * 64];

  // stage packed fragments (uint4 = 8 halves per copy)
  {
    const uint4* g1 = (const uint4*)(w1p + (size_t)layer * 2048);
    const uint4* g2 = (const uint4*)(w2h + (size_t)layer * 4096);
    const uint4* g3 = (const uint4*)(w3h + (size_t)layer * 10240);
    uint4* s1 = (uint4*)w1s;
    uint4* s2 = (uint4*)w2s;
    uint4* s3 = (uint4*)w3s;
    for (int i = threadIdx.x; i < 2048 / 8; i += 256)  s1[i] = g1[i];
    for (int i = threadIdx.x; i < 4096 / 8; i += 256)  s2[i] = g2[i];
    for (int i = threadIdx.x; i < 10240 / 8; i += 256) s3[i] = g3[i];
  }
  __syncthreads();

  const int wave = threadIdx.x >> 5;
  const int lane = threadIdx.x & 31;
  const int n16  = lane & 15;
  const int kh   = lane >> 4;
  const int mb   = kh << 3;           // this lane's D-layout row base (8 edges)
  _Float16* hw = hbuf + wave * 16 * 64;

  const int gwave  = blockIdx.x * 8 + wave;
  const int nwaves = gridDim.x * 8;

  for (int tile = gwave; tile < E_TILES; tile += nwaves) {
    const int e0 = tile * 16;

    // ---- GEMM1: rb[16x8 padded to K=32] x w1[32x64] ----
    v8h rlo = {};
    if (kh == 0)
      rlo = *(const v8h*)(rb16 + (size_t)(e0 + n16) * NBASIS);  // one b128, 16B aligned
    v8h zer = {};
    v16h a1f = __builtin_shufflevector(rlo, zer, 0, 1, 2, 3, 4, 5, 6, 7,
                                                 8, 9, 10, 11, 12, 13, 14, 15);
    if (tile + nwaves < E_TILES)
      __builtin_prefetch(rb16 + (size_t)(tile + nwaves) * 16 * NBASIS, 0, 1);

#pragma unroll
    for (int t = 0; t < 4; t++) {
      v8f c = {};
      c = __builtin_amdgcn_wmma_f32_16x16x32_f16(false, a1f, false,
              loadBfrag(w1s, t, lane), (short)0, c, false, false);
#pragma unroll
      for (int r = 0; r < 8; r++)
        hw[(mb + r) * 64 + t * 16 + n16] = (_Float16)silu_f(c[r]);
    }

    // ---- GEMM2: h1[16x64] x w2[64x64] ----
    v16h a0 = loadA32(hw, 0, lane);
    v16h a1 = loadA32(hw, 1, lane);
    float h2[4][8];
#pragma unroll
    for (int t = 0; t < 4; t++) {
      v8f c = {};
      c = __builtin_amdgcn_wmma_f32_16x16x32_f16(false, a0, false,
              loadBfrag(w2s, t * 2 + 0, lane), (short)0, c, false, false);
      c = __builtin_amdgcn_wmma_f32_16x16x32_f16(false, a1, false,
              loadBfrag(w2s, t * 2 + 1, lane), (short)0, c, false, false);
#pragma unroll
      for (int r = 0; r < 8; r++) h2[t][r] = silu_f(c[r]);
    }
#pragma unroll
    for (int t = 0; t < 4; t++)
#pragma unroll
      for (int r = 0; r < 8; r++)
        hw[(mb + r) * 64 + t * 16 + n16] = (_Float16)h2[t][r];

    a0 = loadA32(hw, 0, lane);
    a1 = loadA32(hw, 1, lane);

    // ---- preload per-edge gather data (shared across all 10 output tiles) ----
    int   rcv8[8];
    float nh8[8][3];
    float su8[2][8];      // s[sender][ch] for ch = h*16 + n16
    float vv8[2][8][3];   // v[sender][ch][d]
    float ndv8[2][8];     // <nhat, v[sender][ch]>
#pragma unroll
    for (int r = 0; r < 8; r++) {
      int e  = e0 + mb + r;
      int sd = senders[e];
      rcv8[r] = receivers[e];
#pragma unroll
      for (int d = 0; d < 3; d++) nh8[r][d] = nhat[(size_t)e * 3 + d];
#pragma unroll
      for (int h = 0; h < 2; h++) {
        int ch = h * 16 + n16;
        su8[h][r] = s_in[(size_t)sd * CCH + ch];
        float dd = 0.f;
#pragma unroll
        for (int d = 0; d < 3; d++) {
          float vd = v_in[((size_t)sd * CCH + ch) * 3 + d];
          vv8[h][r][d] = vd;
          dd += vd * nh8[r][d];
        }
        ndv8[h][r] = dd;
      }
    }

    // ---- GEMM3: h2[16x64] x w3[64x160], fused path scatter per 16-col tile ----
#pragma unroll
    for (int t = 0; t < 10; t++) {
      v8f c = {};
      c = __builtin_amdgcn_wmma_f32_16x16x32_f16(false, a0, false,
              loadBfrag(w3s, t * 2 + 0, lane), (short)0, c, false, false);
      c = __builtin_amdgcn_wmma_f32_16x16x32_f16(false, a1, false,
              loadBfrag(w3s, t * 2 + 1, lane), (short)0, c, false, false);
      const int path = t >> 1;           // tile lies fully inside one path block
      const int hh   = t & 1;
      const int ch   = hh * 16 + n16;
#pragma unroll
      for (int r = 0; r < 8; r++) {
        float w  = c[r];
        int  rcv = rcv8[r];
        if (path == 0) {                 // 0e x Y0 -> 0e
          atomicAdd(&agg_s[(size_t)rcv * 64 + ch], w * su8[hh][r]);
        } else if (path == 1) {          // 1o x Y1 -> 0e
          atomicAdd(&agg_s[(size_t)rcv * 64 + 32 + ch], w * ndv8[hh][r]);
        } else if (path == 2) {          // 0e x Y1 -> 1o
          float su = su8[hh][r];
#pragma unroll
          for (int d = 0; d < 3; d++)
            atomicAdd(&agg_v[((size_t)rcv * 96 + ch) * 3 + d], w * su * nh8[r][d]);
        } else if (path == 3) {          // 1o x Y0 -> 1o
#pragma unroll
          for (int d = 0; d < 3; d++)
            atomicAdd(&agg_v[((size_t)rcv * 96 + 32 + ch) * 3 + d], w * vv8[hh][r][d]);
        } else {                         // 1o x Y2 -> 1o
          float nd = ndv8[hh][r];
#pragma unroll
          for (int d = 0; d < 3; d++)
            atomicAdd(&agg_v[((size_t)rcv * 96 + 64 + ch) * 3 + d],
                      w * (nh8[r][d] * nd - vv8[hh][r][d] * (1.f / 3.f)));
        }
      }
    }
  }
}

// ---- node kernel: linears + per-species self-connection + gated activation ----
__global__ __launch_bounds__(256) void node_kernel(
    int layer, const float* __restrict__ agg_s, const float* __restrict__ agg_v,
    const float* __restrict__ s_in, const float* __restrict__ v_in,
    const int* __restrict__ spec,
    const float* __restrict__ w_lin_s, const float* __restrict__ w_lin_v,
    const float* __restrict__ w_sc_s, const float* __restrict__ w_sc_v,
    float* __restrict__ s_out, float* __restrict__ v_out) {
  __shared__ float wls[64 * 64];
  __shared__ float wlv[96 * 32];
  for (int i = threadIdx.x; i < 64 * 64; i += 256) wls[i] = w_lin_s[(size_t)layer * 64 * 64 + i];
  for (int i = threadIdx.x; i < 96 * 32; i += 256) wlv[i] = w_lin_v[(size_t)layer * 96 * 32 + i];
  __syncthreads();

  int node = blockIdx.x * 8 + (threadIdx.x >> 5);
  int lane = threadIdx.x & 31;
  if (node >= N_NODES) return;

  int sp = spec[node];
  const float* scs = w_sc_s + ((size_t)(layer * NSPEC + sp)) * 32 * 64;
  const float* scv = w_sc_v + ((size_t)(layer * NSPEC + sp)) * 32 * 32;

  const float* as = agg_s + (size_t)node * 64;
  float sg0 = 0.f, sg1 = 0.f;
#pragma unroll 8
  for (int j = 0; j < 64; j++) {
    float av = as[j] * (1.f / 16.f);             // /AVG_NEIGH
    sg0 += av * wls[j * 64 + lane];
    sg1 += av * wls[j * 64 + 32 + lane];
  }
  const float* snp = s_in + (size_t)node * 32;
#pragma unroll 8
  for (int c = 0; c < 32; c++) {
    float sv = snp[c];
    sg0 += sv * scs[c * 64 + lane];
    sg1 += sv * scs[c * 64 + 32 + lane];
  }
  float vn0 = 0.f, vn1 = 0.f, vn2 = 0.f;
  const float* av3 = agg_v + (size_t)node * 96 * 3;
#pragma unroll 8
  for (int p = 0; p < 96; p++) {
    float wv = wlv[p * 32 + lane] * (1.f / 16.f);
    vn0 += av3[p * 3 + 0] * wv;
    vn1 += av3[p * 3 + 1] * wv;
    vn2 += av3[p * 3 + 2] * wv;
  }
  const float* vnp = v_in + (size_t)node * 96;   // [32][3]
#pragma unroll 8
  for (int c = 0; c < 32; c++) {
    float wv = scv[c * 32 + lane];
    vn0 += vnp[c * 3 + 0] * wv;
    vn1 += vnp[c * 3 + 1] * wv;
    vn2 += vnp[c * 3 + 2] * wv;
  }
  s_out[(size_t)node * 32 + lane] = silu_f(sg0);
  float g = silu_f(sg1);
  float* vo = v_out + ((size_t)node * 32 + lane) * 3;
  vo[0] = g * vn0;
  vo[1] = g * vn1;
  vo[2] = g * vn2;
}

__global__ void out_kernel(const float* __restrict__ s, const float* __restrict__ w1,
                           const float* __restrict__ w2, float* __restrict__ out) {
  int n = blockIdx.x * blockDim.x + threadIdx.x;
  if (n >= N_NODES) return;
  const float* sp = s + (size_t)n * 32;
  float acc = 0.f;
#pragma unroll
  for (int k = 0; k < 16; k++) {
    float h = 0.f;
#pragma unroll
    for (int c = 0; c < 32; c++) h += sp[c] * w1[c * 16 + k];
    acc += h * w2[k];
  }
  out[n] = acc;
}

extern "C" void kernel_launch(void* const* d_in, const int* in_sizes, int n_in,
                              void* d_out, int out_size, void* d_ws, size_t ws_size,
                              hipStream_t stream) {
  (void)in_sizes; (void)n_in; (void)out_size; (void)ws_size;

  const float* vectors   = (const float*)d_in[0];
  const int*   spec      = (const int*)  d_in[1];
  const int*   senders   = (const int*)  d_in[2];
  const int*   receivers = (const int*)  d_in[3];
  const float* emb       = (const float*)d_in[4];
  const float* w_rad1    = (const float*)d_in[5];
  const float* w_rad2    = (const float*)d_in[6];
  const float* w_rad_out = (const float*)d_in[7];
  const float* w_lin_s   = (const float*)d_in[8];
  const float* w_lin_v   = (const float*)d_in[9];
  const float* w_sc_s    = (const float*)d_in[10];
  const float* w_sc_v    = (const float*)d_in[11];
  const float* w_out1    = (const float*)d_in[12];
  const float* w_out2    = (const float*)d_in[13];

  char*  ws  = (char*)d_ws;
  size_t off = 0;
  auto take = [&](size_t bytes) -> char* {
    char* p = ws + off;
    off += (bytes + 255) & ~(size_t)255;
    return p;
  };
  _Float16* rb16 = (_Float16*)take((size_t)N_EDGES * NBASIS * 2);
  float*    nhat = (float*)   take((size_t)N_EDGES * 3 * 4);
  _Float16* w1p  = (_Float16*)take((size_t)LLAYERS * 2048 * 2);
  _Float16* w2h  = (_Float16*)take((size_t)LLAYERS * 4096 * 2);
  _Float16* w3h  = (_Float16*)take((size_t)LLAYERS * 10240 * 2);
  float*    sA   = (float*)   take((size_t)N_NODES * 32 * 4);
  float*    sB   = (float*)   take((size_t)N_NODES * 32 * 4);
  float*    vA   = (float*)   take((size_t)N_NODES * 96 * 4);
  float*    vB   = (float*)   take((size_t)N_NODES * 96 * 4);
  float*    aggs = (float*)   take((size_t)N_NODES * 64 * 4);
  float*    aggv = (float*)   take((size_t)N_NODES * 288 * 4);

  prep_edges<<<(N_EDGES + 255) / 256, 256, 0, stream>>>(vectors, rb16, nhat);
  {
    const int total = LLAYERS * (2048 + 4096 + 10240);
    convert_weights<<<(total + 255) / 256, 256, 0, stream>>>(w_rad1, w_rad2, w_rad_out,
                                                             w1p, w2h, w3h);
  }
  init_nodes<<<(N_NODES * CCH + 255) / 256, 256, 0, stream>>>(emb, spec, sA);
  hipMemsetAsync(vA, 0, (size_t)N_NODES * 96 * 4, stream);

  float* sc = sA; float* vc = vA;
  float* sn = sB; float* vn = vB;
  for (int l = 0; l < LLAYERS; l++) {
    hipMemsetAsync(aggs, 0, (size_t)N_NODES * 64 * 4, stream);
    hipMemsetAsync(aggv, 0, (size_t)N_NODES * 288 * 4, stream);
    edge_kernel<<<469, 256, 0, stream>>>(l, rb16, nhat, senders, receivers, sc, vc,
                                         w1p, w2h, w3h, aggs, aggv);
    node_kernel<<<(N_NODES + 7) / 8, 256, 0, stream>>>(l, aggs, aggv, sc, vc, spec,
                                                       w_lin_s, w_lin_v, w_sc_s, w_sc_v,
                                                       sn, vn);
    float* ts = sc; sc = sn; sn = ts;
    float* tv = vc; vc = vn; vn = tv;
  }
  out_kernel<<<(N_NODES + 255) / 256, 256, 0, stream>>>(sc, w_out1, w_out2, (float*)d_out);
}
